// Attention_78761110274699
// MI455X (gfx1250) — matrix operand order repaired and checked by reading, not compiled, not run on hardware
//
#include <hip/hip_runtime.h>
#include <hip/hip_bf16.h>
#include <math.h>

typedef __attribute__((ext_vector_type(16))) _Float16 v16h;
typedef __attribute__((ext_vector_type(8)))  _Float16 v8h;
typedef __attribute__((ext_vector_type(2)))  __fp16   v2hf;   // pkrtz result type
typedef __attribute__((ext_vector_type(8)))  float    v8f;
typedef __attribute__((ext_vector_type(4)))  int      v4i;

#define B_  2
#define N_  2048
#define D_  1024
#define H_  16
#define HD_ 64
#define EPS_ 1e-5f

#if __has_builtin(__builtin_amdgcn_global_load_async_to_lds_b128) && \
    __has_builtin(__builtin_amdgcn_s_wait_asynccnt)
#define USE_ASYNC_LDS 1
typedef __attribute__((address_space(1))) v4i GV4;
typedef __attribute__((address_space(3))) v4i LV4;
#else
#define USE_ASYNC_LDS 0
#endif

// ---------------------------------------------------------------------------
// Helpers
// ---------------------------------------------------------------------------
__device__ inline v8f wmma16x16x32(v16h a, v16h b, v8f c) {
  return __builtin_amdgcn_wmma_f32_16x16x32_f16(
      false, a, false, b, (short)0, c, false, false);
}

// 8 floats -> 8 f16 via v_cvt_pk_rtz_f16_f32, bit-assembled.
__device__ inline v8h pack8(float4 a, float4 b) {
  union { v2hf p[4]; v8h v; } u;
  u.p[0] = __builtin_amdgcn_cvt_pkrtz(a.x, a.y);
  u.p[1] = __builtin_amdgcn_cvt_pkrtz(a.z, a.w);
  u.p[2] = __builtin_amdgcn_cvt_pkrtz(b.x, b.y);
  u.p[3] = __builtin_amdgcn_cvt_pkrtz(b.z, b.w);
  return u.v;
}

// A-matrix 16x32 f16 fragment from row-major storage.
__device__ inline v16h load_a_frag(const _Float16* src, int row_stride) {
  int lane = threadIdx.x & 31;
  const _Float16* p = src + (lane & 15) * row_stride + (lane >> 4) * 8;
  v16h a;
#pragma unroll
  for (int j = 0; j < 8; ++j) {
    a[j]     = p[j];
    a[8 + j] = p[16 + j];
  }
  return a;
}

// B-matrix 32x16 fragment, element (k,n) = src[n*row_stride + k]
__device__ inline v16h load_b_frag(const _Float16* src, int row_stride) {
  int lane = threadIdx.x & 31;
  const _Float16* p = src + (lane & 15) * row_stride + (lane >> 4) * 8;
  v16h b;
#pragma unroll
  for (int j = 0; j < 8; ++j) {
    b[j]     = p[j];
    b[8 + j] = p[16 + j];
  }
  return b;
}

// ---------------------------------------------------------------------------
// GEMM: C[M x NOUT] = X[M x K] @ W[NOUT x K]^T + bias
// TIN: float (cvt on stage) or _Float16 (raw copy); TOUT: float or _Float16.
// 8 waves; block tile 128x64; wave tile 32x32 (4 WMMA); double-buffered LDS.
// ---------------------------------------------------------------------------
#define GBM 128
#define GBN 64
#define GBK 32
template <typename TIN, typename TOUT>
__global__ __launch_bounds__(256) void gemm_xwT_bias(
    const TIN* __restrict__ X, const float* __restrict__ W,
    const float* __restrict__ bias, TOUT* __restrict__ C,
    int M, int NOUT, int K)
{
  __shared__ _Float16 sA[2][GBM][GBK];   // 16 KB
  __shared__ _Float16 sB[2][GBN][GBK];   // 8 KB
  const int tid  = threadIdx.x;
  const int wave = tid >> 5;
  const int lane = tid & 31;
  const int wm   = wave & 3;
  const int wn   = wave >> 2;
  const int m0 = blockIdx.x * GBM;
  const int n0 = blockIdx.y * GBN;

  const int arow = tid >> 1, ac0 = (tid & 1) * 16;
  const int brow = tid >> 2, bc0 = (tid & 3) * 8;
  const TIN*   Ap = X + (size_t)(m0 + arow) * K + ac0;
  const float* Bp = W + (size_t)(n0 + brow) * K + bc0;

  float4 aRf[4], bRf[2];
  v8h    aRh[2];
  auto load_regs = [&](int k0) {
    if constexpr (sizeof(TIN) == 4) {
      const float4* ap4 = (const float4*)(Ap + k0);
      aRf[0] = ap4[0]; aRf[1] = ap4[1]; aRf[2] = ap4[2]; aRf[3] = ap4[3];
    } else {
      const v8h* ap8 = (const v8h*)(Ap + k0);
      aRh[0] = ap8[0]; aRh[1] = ap8[1];
    }
    const float4* bp4 = (const float4*)(Bp + k0);
    bRf[0] = bp4[0]; bRf[1] = bp4[1];
  };
  auto store_lds = [&](int buf) {
    v8h* da = (v8h*)&sA[buf][arow][ac0];
    if constexpr (sizeof(TIN) == 4) {
      da[0] = pack8(aRf[0], aRf[1]);
      da[1] = pack8(aRf[2], aRf[3]);
    } else {
      da[0] = aRh[0];
      da[1] = aRh[1];
    }
    *(v8h*)&sB[buf][brow][bc0] = pack8(bRf[0], bRf[1]);
  };

  load_regs(0);
  store_lds(0);
  __syncthreads();

  v8f acc00 = {}, acc01 = {}, acc10 = {}, acc11 = {};
  const int nk = K / GBK;
  for (int i = 0; i < nk; ++i) {
    const int cur = i & 1;
    if (i + 1 < nk) load_regs((i + 1) * GBK);
    v16h a0 = load_a_frag(&sA[cur][wm * 32][0],      GBK);
    v16h a1 = load_a_frag(&sA[cur][wm * 32 + 16][0], GBK);
    v16h b0 = load_b_frag(&sB[cur][wn * 32][0],      GBK);
    v16h b1 = load_b_frag(&sB[cur][wn * 32 + 16][0], GBK);
    acc00 = wmma16x16x32(a0, b0, acc00);
    acc01 = wmma16x16x32(a0, b1, acc01);
    acc10 = wmma16x16x32(a1, b0, acc10);
    acc11 = wmma16x16x32(a1, b1, acc11);
    if (i + 1 < nk) store_lds(cur ^ 1);
    __syncthreads();
  }

  const int colb = n0 + wn * 32 + (lane & 15);
  const int rowb = m0 + wm * 32 + (lane >> 4) * 8;
  const float bv0 = bias[colb];
  const float bv1 = bias[colb + 16];
#pragma unroll
  for (int r = 0; r < 8; ++r) {
    C[(size_t)(rowb + r)      * NOUT + colb]      = (TOUT)(acc00[r] + bv0);
    C[(size_t)(rowb + r)      * NOUT + colb + 16] = (TOUT)(acc01[r] + bv1);
    C[(size_t)(rowb + 16 + r) * NOUT + colb]      = (TOUT)(acc10[r] + bv0);
    C[(size_t)(rowb + 16 + r) * NOUT + colb + 16] = (TOUT)(acc11[r] + bv1);
  }
}

// ---------------------------------------------------------------------------
// Per-head LayerNorm over contiguous rows of 64 f16; f32 statistics.
// ---------------------------------------------------------------------------
__global__ __launch_bounds__(256) void head_layernorm(
    _Float16* __restrict__ X, const float* __restrict__ gamma,
    const float* __restrict__ beta, int nrows)
{
  const int wave = threadIdx.x >> 5;
  const int lane = threadIdx.x & 31;
  const int row  = blockIdx.x * 8 + wave;
  if (row >= nrows) return;
  _Float16* p = X + (size_t)row * HD_;
  float x0 = (float)p[lane];
  float x1 = (float)p[32 + lane];
  float s  = x0 + x1;
  float s2 = x0 * x0 + x1 * x1;
#pragma unroll
  for (int m = 16; m >= 1; m >>= 1) {
    s  += __shfl_xor(s,  m, 32);
    s2 += __shfl_xor(s2, m, 32);
  }
  float mu   = s * (1.0f / HD_);
  float var  = s2 * (1.0f / HD_) - mu * mu;
  float rstd = rsqrtf(var + EPS_);
  p[lane]      = (_Float16)((x0 - mu) * rstd * gamma[lane]      + beta[lane]);
  p[32 + lane] = (_Float16)((x1 - mu) * rstd * gamma[32 + lane] + beta[32 + lane]);
}

// ---------------------------------------------------------------------------
// Flash attention on f16 Q/K/V: block = (b, h, 128-query slab); 8 waves x 16 q.
// K chunk staged via GLOBAL_LOAD_ASYNC_TO_LDS_B128 (fallback: reg copy);
// V staged transposed so P@V B-fragments are contiguous LDS reads.
// ---------------------------------------------------------------------------
#define AT_QW 16
#define AT_KC 32
__global__ __launch_bounds__(256) void flash_attn(
    const _Float16* __restrict__ Q, const _Float16* __restrict__ Km,
    const _Float16* __restrict__ V, _Float16* __restrict__ O)
{
  __shared__ _Float16 sK[2][AT_KC][HD_];    // 8 KB (row-major: key x hd)
  __shared__ _Float16 sVt[2][HD_][AT_KC];   // 8 KB (transposed: hd x key)
  __shared__ _Float16 sP[8][AT_QW * AT_KC]; // 8 KB (wave-private P staging)

  const int tid  = threadIdx.x;
  const int wave = tid >> 5;
  const int lane = tid & 31;
  const int qb   = blockIdx.x & 15;
  const int h    = (blockIdx.x >> 4) & 15;
  const int b    = blockIdx.x >> 8;
  const int q0   = qb * 128 + wave * AT_QW;

  // Q fragments (16x64 -> two 16x32 k-steps), raw f16 vector loads.
  const _Float16* qp = Q + (size_t)(b * N_ + q0 + (lane & 15)) * D_ + h * HD_
                         + (lane >> 4) * 8;
  union { v16h v; v8h h[2]; } uq0, uq1;
  uq0.h[0] = *(const v8h*)(qp);
  uq0.h[1] = *(const v8h*)(qp + 16);
  uq1.h[0] = *(const v8h*)(qp + 32);
  uq1.h[1] = *(const v8h*)(qp + 48);
  const v16h aq0 = uq0.v, aq1 = uq1.v;

  // Staging slice: thread -> 8 contiguous f16 of one key row (16 bytes).
  const int krow = tid >> 3, kc0 = (tid & 7) * 8;
  const _Float16* Kp = Km + (size_t)(b * N_ + krow) * D_ + h * HD_ + kc0;
  const _Float16* Vp = V  + (size_t)(b * N_ + krow) * D_ + h * HD_ + kc0;

  v8h vH;
#if USE_ASYNC_LDS
#define STAGE_K(kb, buf)                                                    \
  __builtin_amdgcn_global_load_async_to_lds_b128(                           \
      (GV4*)(Kp + (size_t)(kb) * D_),                                       \
      (LV4*)&sK[buf][krow][kc0], 0, 0)
#define WAIT_K() __builtin_amdgcn_s_wait_asynccnt(0)
#else
  v8h kH;
#define STAGE_K(kb, buf)                                                    \
  do {                                                                      \
    kH = *(const v8h*)(Kp + (size_t)(kb) * D_);                             \
    *(v8h*)&sK[buf][krow][kc0] = kH;                                        \
  } while (0)
#define WAIT_K() ((void)0)
#endif
#define LOAD_V(kb)  (vH = *(const v8h*)(Vp + (size_t)(kb) * D_))
#define STORE_VT(buf)                                                       \
  do {                                                                      \
    _Float16* dv = &sVt[buf][kc0][krow];                                    \
    _Pragma("unroll")                                                       \
    for (int j = 0; j < 8; ++j) dv[j * AT_KC] = vH[j];                      \
  } while (0)

  STAGE_K(0, 0);
  LOAD_V(0);
  STORE_VT(0);
  WAIT_K();
  __syncthreads();

  v8f o0 = {}, o1 = {}, o2 = {}, o3 = {};
  float mstat[8], lstat[8];
#pragma unroll
  for (int r = 0; r < 8; ++r) { mstat[r] = -1e30f; lstat[r] = 0.0f; }

  const float scale = 0.125f;  // 1/sqrt(64)
  const int nchunk = N_ / AT_KC;
  for (int ic = 0; ic < nchunk; ++ic) {
    const int cur = ic & 1;
    if (ic + 1 < nchunk) {
      STAGE_K((ic + 1) * AT_KC, cur ^ 1);
      LOAD_V((ic + 1) * AT_KC);
    }

    // S = Q @ K^T (two 16-key subtiles; key rows are n-major in sK)
    v8f s0 = {}, s1 = {};
    s0 = wmma16x16x32(aq0, load_b_frag(&sK[cur][0][0],   HD_), s0);
    s0 = wmma16x16x32(aq1, load_b_frag(&sK[cur][0][32],  HD_), s0);
    s1 = wmma16x16x32(aq0, load_b_frag(&sK[cur][16][0],  HD_), s1);
    s1 = wmma16x16x32(aq1, load_b_frag(&sK[cur][16][32], HD_), s1);

    // Online softmax (rows r / r+8 owned by lane halves; xor<=8 stays in half)
    _Float16* pb = &sP[wave][0];
#pragma unroll
    for (int r = 0; r < 8; ++r) {
      float v0 = s0[r] * scale;
      float v1 = s1[r] * scale;
      float mx = fmaxf(v0, v1);
#pragma unroll
      for (int m = 8; m >= 1; m >>= 1) mx = fmaxf(mx, __shfl_xor(mx, m, 32));
      float mnew = fmaxf(mstat[r], mx);
      float corr = __expf(mstat[r] - mnew);
      float p0 = __expf(v0 - mnew);
      float p1 = __expf(v1 - mnew);
      float rs = p0 + p1;
#pragma unroll
      for (int m = 8; m >= 1; m >>= 1) rs += __shfl_xor(rs, m, 32);
      lstat[r] = lstat[r] * corr + rs;
      mstat[r] = mnew;
      o0[r] *= corr; o1[r] *= corr; o2[r] *= corr; o3[r] *= corr;
      int prow = r + (lane >> 4) * 8;
      pb[prow * AT_KC + (lane & 15)]      = (_Float16)p0;
      pb[prow * AT_KC + 16 + (lane & 15)] = (_Float16)p1;
    }

    // O += P(16x32) @ V(32x64); sVt rows are output-cols -> contiguous frags.
    v16h ap = load_a_frag(pb, AT_KC);
    o0 = wmma16x16x32(ap, load_b_frag(&sVt[cur][0][0],  AT_KC), o0);
    o1 = wmma16x16x32(ap, load_b_frag(&sVt[cur][16][0], AT_KC), o1);
    o2 = wmma16x16x32(ap, load_b_frag(&sVt[cur][32][0], AT_KC), o2);
    o3 = wmma16x16x32(ap, load_b_frag(&sVt[cur][48][0], AT_KC), o3);

    if (ic + 1 < nchunk) STORE_VT(cur ^ 1);
    WAIT_K();
    __syncthreads();
  }
#undef STAGE_K
#undef WAIT_K
#undef LOAD_V
#undef STORE_VT

  // Normalize and write O as f16 (layout [B,N,D])
#pragma unroll
  for (int r = 0; r < 8; ++r) {
    float inv = 1.0f / lstat[r];
    int qrow = q0 + (lane >> 4) * 8 + r;
    _Float16* op = O + (size_t)(b * N_ + qrow) * D_ + h * HD_ + (lane & 15);
    op[0]  = (_Float16)(o0[r] * inv);
    op[16] = (_Float16)(o1[r] * inv);
    op[32] = (_Float16)(o2[r] * inv);
    op[48] = (_Float16)(o3[r] * inv);
  }
}

// ---------------------------------------------------------------------------
// Launch
// ---------------------------------------------------------------------------
extern "C" void kernel_launch(void* const* d_in, const int* in_sizes, int n_in,
                              void* d_out, int out_size, void* d_ws, size_t ws_size,
                              hipStream_t stream) {
  const float* query   = (const float*)d_in[0];
  const float* Wq      = (const float*)d_in[1];
  const float* bq      = (const float*)d_in[2];
  const float* Wk      = (const float*)d_in[3];
  const float* bk      = (const float*)d_in[4];
  const float* Wv      = (const float*)d_in[5];
  const float* bv      = (const float*)d_in[6];
  const float* Wp      = (const float*)d_in[7];
  const float* bp      = (const float*)d_in[8];
  const float* q_gamma = (const float*)d_in[9];
  const float* q_beta  = (const float*)d_in[10];
  const float* k_gamma = (const float*)d_in[11];
  const float* k_beta  = (const float*)d_in[12];

  const size_t TOK = (size_t)B_ * N_ * D_;     // 4M f16 elements per buffer
  _Float16* Qh = (_Float16*)d_ws;
  _Float16* Kh = Qh + TOK;
  _Float16* Vh = Kh + TOK;
  _Float16* Oh = Vh + TOK;

  const int M = B_ * N_;                       // 4096
  dim3 gemm_grid(M / GBM, D_ / GBN);           // (32, 16)

  gemm_xwT_bias<float, _Float16><<<gemm_grid, 256, 0, stream>>>(query, Wq, bq, Qh, M, D_, D_);
  gemm_xwT_bias<float, _Float16><<<gemm_grid, 256, 0, stream>>>(query, Wk, bk, Kh, M, D_, D_);
  gemm_xwT_bias<float, _Float16><<<gemm_grid, 256, 0, stream>>>(query, Wv, bv, Vh, M, D_, D_);

  const int nrows = B_ * N_ * H_;              // 65536
  head_layernorm<<<nrows / 8, 256, 0, stream>>>(Qh, q_gamma, q_beta, nrows);
  head_layernorm<<<nrows / 8, 256, 0, stream>>>(Kh, k_gamma, k_beta, nrows);

  flash_attn<<<B_ * H_ * (N_ / 128), 256, 0, stream>>>(Qh, Kh, Vh, Oh);

  gemm_xwT_bias<_Float16, float><<<gemm_grid, 256, 0, stream>>>(Oh, Wp, bp, (float*)d_out, M, D_, D_);
}
